// GATClassifier_25348896981474
// MI455X (gfx1250) — matrix operand order repaired and checked by reading, not compiled, not run on hardware
//
#include <hip/hip_runtime.h>
#include <hip/hip_bf16.h>
#include <cstddef>

// ---------------------------------------------------------------------------
// Types for CDNA5 WMMA (wave32, 16x16x32 bf16 -> f32 accumulate)
// ---------------------------------------------------------------------------
typedef __attribute__((ext_vector_type(16))) __bf16 v16bf;
typedef __attribute__((ext_vector_type(8)))  float  v8f;
typedef __attribute__((ext_vector_type(2)))  __bf16 v2bf;
typedef __attribute__((ext_vector_type(2)))  float  v2f;

union FragU {
    unsigned int u[8];
    v16bf        v;
};

#define HEADS 4
#define CH    64
#define HC    256   // HEADS*CH
#define NEG_SLOPE 0.2f

// Single v_cvt_pk_bf16_f32: pack two f32 into one dword of 2 bf16.
__device__ __forceinline__ unsigned int pack_bf16(float a, float b) {
    v2f f = {a, b};
    v2bf h = __builtin_convertvector(f, v2bf);
    union { v2bf v; unsigned int u; } t;
    t.v = h;
    return t.u;
}

// ---------------------------------------------------------------------------
// Generic fill
// ---------------------------------------------------------------------------
__global__ void fill_f32(float* __restrict__ p, float v, long n) {
    long i = (long)blockIdx.x * blockDim.x + threadIdx.x;
    if (i < n) p[i] = v;
}

// ---------------------------------------------------------------------------
// bf16 WMMA GEMM:  C[M, 256] = A[M, K] (f32) x B[K, 256] (f32)
// Block tile 128x128 (grid.y = 2 -> x streamed from HBM only twice).
// 8 waves; each wave: 16 rows x 128 cols = 8 wmma accumulators; the A
// fragment is reused across all 8 B tiles. Branchless vectorized staging +
// register prefetch overlap HBM with WMMA. Requires K % 4 == 0 (1300, 256).
// ---------------------------------------------------------------------------
#define BM 128
#define BN 128
#define BK 32
#define NT 8     // BN/16 column tiles per wave
#define APAD 8   // bf16 elements of row padding (keeps dword alignment)

__global__ __launch_bounds__(256)
void gemm_bf16_wmma(const float* __restrict__ A, const float* __restrict__ B,
                    float* __restrict__ C, int M, int K) {
    __shared__ __align__(16) __bf16 As[BM][BK + APAD];   // row-major (m, k)
    __shared__ __align__(16) __bf16 Bs[BN][BK + APAD];   // transposed (n, k)

    const int tid  = threadIdx.x;
    const int wave = tid >> 5;      // 0..7
    const int lane = tid & 31;
    const int rowBase = blockIdx.x * BM;
    const int colBase = blockIdx.y * BN;

    // A staging: 128x32 f32 = 1024 float4; 4 per thread.
    //   float4 index p = tid + i*256 : r = p>>3, c = (p&7)*4
    // B staging: 32x128 f32 = 1024 (2n x 2k) quads; 4 per thread.
    //   quad  index p = tid + i*256 : n2 = (p&63)*2, k = (p>>6)*2
    float4 aReg[4];
    float2 bReg0[4], bReg1[4];

    auto loadTiles = [&](int k0) {
        #pragma unroll
        for (int i = 0; i < 4; ++i) {
            int p  = tid + i * 256;
            int r  = p >> 3, c = (p & 7) * 4;
            int gr = rowBase + r, gc = k0 + c;
            bool ok = (gr < M) && (gc < K);          // float4 never straddles K
            size_t off = ok ? ((size_t)gr * K + gc) : 0;
            float4 v = *(const float4*)(A + off);
            if (!ok) v = make_float4(0.f, 0.f, 0.f, 0.f);
            aReg[i] = v;
        }
        #pragma unroll
        for (int i = 0; i < 4; ++i) {
            int p  = tid + i * 256;
            int n2 = (p & 63) * 2, k = (p >> 6) * 2;
            int gk = k0 + k;
            bool ok = (gk < K);                      // K even -> gk+1 < K too
            size_t off = ok ? ((size_t)gk * HC + colBase + n2) : 0;
            float2 v0 = *(const float2*)(B + off);
            float2 v1 = *(const float2*)(B + off + (ok ? HC : 0));
            if (!ok) { v0 = make_float2(0.f, 0.f); v1 = v0; }
            bReg0[i] = v0; bReg1[i] = v1;
        }
    };

    v8f acc[NT] = {};
    loadTiles(0);

    for (int k0 = 0; k0 < K; k0 += BK) {
        __syncthreads();   // previous step's fragments fully consumed
        // ---- convert + store staged registers into LDS ----
        #pragma unroll
        for (int i = 0; i < 4; ++i) {
            int p = tid + i * 256;
            int r = p >> 3, c = (p & 7) * 4;
            float4 v = aReg[i];
            *(unsigned int*)&As[r][c]     = pack_bf16(v.x, v.y);
            *(unsigned int*)&As[r][c + 2] = pack_bf16(v.z, v.w);
        }
        #pragma unroll
        for (int i = 0; i < 4; ++i) {
            int p = tid + i * 256;
            int n2 = (p & 63) * 2, k = (p >> 6) * 2;
            *(unsigned int*)&Bs[n2][k]     = pack_bf16(bReg0[i].x, bReg1[i].x);
            *(unsigned int*)&Bs[n2 + 1][k] = pack_bf16(bReg0[i].y, bReg1[i].y);
        }
        __syncthreads();

        // ---- prefetch next K-step while WMMAs run ----
        if (k0 + BK < K) loadTiles(k0 + BK);

        // ---- A fragment: 16-bit A 16x32 layout (ISA 7.12.2) ----
        const int mloc  = wave * 16 + (lane & 15);
        const int khalf = (lane >> 4) * 8;
        FragU fa;
        #pragma unroll
        for (int p = 0; p < 8; ++p) {
            int k = ((p & 3) << 1) + ((p >> 2) << 4) + khalf;   // pairs
            fa.u[p] = *(const unsigned int*)&As[mloc][k];
        }

        // ---- B fragments (lane = N, K halves split at lane 16) + WMMA ----
        const int nloc = lane & 15;
        const int kb   = (lane >> 4) * 8;
        #pragma unroll
        for (int ct = 0; ct < NT; ++ct) {
            FragU fb;
            #pragma unroll
            for (int p = 0; p < 8; ++p)
                fb.u[p] = *(const unsigned int*)&Bs[ct * 16 + nloc][kb + ((p & 3) << 1) + ((p >> 2) << 4)];
            acc[ct] = __builtin_amdgcn_wmma_f32_16x16x32_bf16(
                false, fa.v, false, fb.v, (short)0, acc[ct], false, false);
        }
    }

    // ---- epilogue: C/D layout VGPR r -> M = (lane>>4)*8 + r, N = lane&15 ----
    const int rofs = (lane >> 4) * 8;
    const int cofs = lane & 15;
    #pragma unroll
    for (int ct = 0; ct < NT; ++ct) {
        #pragma unroll
        for (int r = 0; r < 8; ++r) {
            int row = rowBase + wave * 16 + rofs + r;
            if (row < M)
                C[(size_t)row * HC + colBase + ct * 16 + cofs] = acc[ct][r];
        }
    }
}

// ---------------------------------------------------------------------------
// a_src[n,h] = <h[n,h,:], att_src[h,:]>,  a_dst likewise. One wave per (n,h).
// ---------------------------------------------------------------------------
__global__ void attn_dots(const float* __restrict__ h,
                          const float* __restrict__ att_s,
                          const float* __restrict__ att_d,
                          float* __restrict__ a_s, float* __restrict__ a_d,
                          int n) {
    int wid  = (int)((blockIdx.x * (long)blockDim.x + threadIdx.x) >> 5);
    int lane = threadIdx.x & 31;
    if (wid >= n * HEADS) return;
    int node = wid >> 2, head = wid & 3;
    const float* hp = h + (size_t)node * HC + head * CH;
    const float* sp = att_s + head * CH;
    const float* dp = att_d + head * CH;
    float s = hp[lane] * sp[lane] + hp[lane + 32] * sp[lane + 32];
    float d = hp[lane] * dp[lane] + hp[lane + 32] * dp[lane + 32];
    #pragma unroll
    for (int m = 16; m; m >>= 1) {
        s += __shfl_xor(s, m, 32);
        d += __shfl_xor(d, m, 32);
    }
    if (lane == 0) { a_s[wid] = s; a_d[wid] = d; }
}

// ---------------------------------------------------------------------------
// Float atomic max via monotonic integer mapping
// ---------------------------------------------------------------------------
__device__ __forceinline__ void atomicMaxFloat(float* addr, float val) {
    if (val >= 0.0f) atomicMax((int*)addr, __float_as_int(val));
    else             atomicMin((unsigned int*)addr, __float_as_uint(val));
}

__device__ __forceinline__ void edge_endpoints(const int* __restrict__ ei,
                                               long eidx, int E,
                                               int& s, int& d) {
    if (eidx < E) { s = ei[eidx]; d = ei[(size_t)E + eidx]; }
    else          { s = d = (int)(eidx - E); }          // self loop
}

__device__ __forceinline__ float leaky(float e) {
    return e > 0.0f ? e : NEG_SLOPE * e;
}

// Pass 1: segment max of leaky logits
__global__ void edge_max(const int* __restrict__ ei,
                         const float* __restrict__ a_s,
                         const float* __restrict__ a_d,
                         float* __restrict__ emax, int E, int n) {
    long idx = (long)blockIdx.x * blockDim.x + threadIdx.x;
    if (idx >= (long)E + n) return;
    int s, d; edge_endpoints(ei, idx, E, s, d);
    #pragma unroll
    for (int h = 0; h < HEADS; ++h) {
        float e = leaky(a_s[s * HEADS + h] + a_d[d * HEADS + h]);
        atomicMaxFloat(&emax[d * HEADS + h], e);
    }
}

// Pass 2: segment sum of exp(e - max)
__global__ void edge_denom(const int* __restrict__ ei,
                           const float* __restrict__ a_s,
                           const float* __restrict__ a_d,
                           const float* __restrict__ emax,
                           float* __restrict__ denom, int E, int n) {
    long idx = (long)blockIdx.x * blockDim.x + threadIdx.x;
    if (idx >= (long)E + n) return;
    int s, d; edge_endpoints(ei, idx, E, s, d);
    #pragma unroll
    for (int h = 0; h < HEADS; ++h) {
        float e = leaky(a_s[s * HEADS + h] + a_d[d * HEADS + h]);
        atomicAdd(&denom[d * HEADS + h], __expf(e - emax[d * HEADS + h]));
    }
}

// Pass 3: alpha-weighted scatter of messages. One thread per (edge, head).
__global__ void edge_scatter(const int* __restrict__ ei,
                             const float* __restrict__ a_s,
                             const float* __restrict__ a_d,
                             const float* __restrict__ emax,
                             const float* __restrict__ denom,
                             const float* __restrict__ hlin,
                             float* __restrict__ agg, int E, int n) {
    long gid = (long)blockIdx.x * blockDim.x + threadIdx.x;
    if (gid >= ((long)E + n) * HEADS) return;
    int  head = (int)(gid & 3);
    long eidx = gid >> 2;
    int s, d; edge_endpoints(ei, eidx, E, s, d);
    float e = leaky(a_s[s * HEADS + head] + a_d[d * HEADS + head]);
    float alpha = __expf(e - emax[d * HEADS + head]) /
                  (denom[d * HEADS + head] + 1e-16f);
    const float4* hs = (const float4*)(hlin + (size_t)s * HC + head * CH);
    float* op = agg + (size_t)d * HC + head * CH;
    #pragma unroll
    for (int i = 0; i < CH / 4; ++i) {
        float4 v = hs[i];
        atomicAdd(op + i * 4 + 0, v.x * alpha);
        atomicAdd(op + i * 4 + 1, v.y * alpha);
        atomicAdd(op + i * 4 + 2, v.z * alpha);
        atomicAdd(op + i * 4 + 3, v.w * alpha);
    }
}

__global__ void bias_relu(const float* __restrict__ agg,
                          const float* __restrict__ bias,
                          float* __restrict__ out, long n) {
    long i = (long)blockIdx.x * blockDim.x + threadIdx.x;
    if (i >= n) return;
    float v = agg[i] + bias[i & (HC - 1)];
    out[i] = v > 0.0f ? v : 0.0f;
}

// ---------------------------------------------------------------------------
// Global mean pool (atomic accumulate) + final linear head
// ---------------------------------------------------------------------------
__global__ void pool_accum(const float* __restrict__ h,
                           const int* __restrict__ batch,
                           float* __restrict__ pooled,
                           float* __restrict__ cnt, int n) {
    long t = (long)blockIdx.x * blockDim.x + threadIdx.x;
    int node = (int)(t >> 2), q = (int)(t & 3);
    if (node >= n) return;
    int g = batch[node];
    if (q == 0) atomicAdd(&cnt[g], 1.0f);
    const float* hp = h + (size_t)node * HC + q * 64;
    float* pp = pooled + (size_t)g * HC + q * 64;
    #pragma unroll
    for (int i = 0; i < 64; ++i) atomicAdd(pp + i, hp[i]);
}

__global__ void final_lin(const float* __restrict__ pooled,
                          const float* __restrict__ cnt,
                          const float* __restrict__ w,
                          const float* __restrict__ b,
                          float* __restrict__ out, int g_) {
    int wid  = (int)((blockIdx.x * (long)blockDim.x + threadIdx.x) >> 5);
    int lane = threadIdx.x & 31;
    if (wid >= g_) return;
    const float* pp = pooled + (size_t)wid * HC;
    float s = 0.0f;
    #pragma unroll
    for (int j = 0; j < HC / 32; ++j) s += pp[lane + j * 32] * w[lane + j * 32];
    #pragma unroll
    for (int m = 16; m; m >>= 1) s += __shfl_xor(s, m, 32);
    if (lane == 0) out[wid] = s / fmaxf(cnt[wid], 1.0f) + b[0];
}

// ---------------------------------------------------------------------------
// Host-side orchestration
// ---------------------------------------------------------------------------
static inline int blocks_for(long n, int b) { return (int)((n + b - 1) / b); }

extern "C" void kernel_launch(void* const* d_in, const int* in_sizes, int n_in,
                              void* d_out, int out_size, void* d_ws, size_t ws_size,
                              hipStream_t stream) {
    const float* x    = (const float*)d_in[0];
    const int*   ei   = (const int*)d_in[1];
    const int*   batc = (const int*)d_in[2];
    const float* W1   = (const float*)d_in[3];
    const float* as1  = (const float*)d_in[4];
    const float* ad1  = (const float*)d_in[5];
    const float* b1   = (const float*)d_in[6];
    const float* W2   = (const float*)d_in[7];
    const float* as2  = (const float*)d_in[8];
    const float* ad2  = (const float*)d_in[9];
    const float* b2   = (const float*)d_in[10];
    const float* lw   = (const float*)d_in[11];
    const float* lb   = (const float*)d_in[12];

    const int N  = in_sizes[2];
    const int E  = in_sizes[1] / 2;
    const int Kin = in_sizes[0] / N;     // 1300
    const int G  = out_size;             // 64
    float* out = (float*)d_out;

    // workspace carve-up
    float* ws = (float*)d_ws;
    size_t o = 0;
    float* hlin  = ws + o; o += (size_t)N * HC;
    float* h1    = ws + o; o += (size_t)N * HC;
    float* agg   = ws + o; o += (size_t)N * HC;
    float* a_s   = ws + o; o += (size_t)N * HEADS;
    float* a_d   = ws + o; o += (size_t)N * HEADS;
    float* emax  = ws + o; o += (size_t)N * HEADS;
    float* denom = ws + o; o += (size_t)N * HEADS;
    float* pooled= ws + o; o += (size_t)G * HC;
    float* cnt   = ws + o; o += (size_t)G;
    (void)ws_size; (void)n_in;

    const long nh   = (long)N * HEADS;
    const long nhc  = (long)N * HC;
    const long etot = (long)E + N;

    auto layer = [&](const float* xin, int Kdim, const float* W,
                     const float* att_s, const float* att_d,
                     const float* bias, float* hout) {
        // h = x @ W  (bf16 WMMA)
        dim3 grid((N + BM - 1) / BM, HC / BN);
        gemm_bf16_wmma<<<grid, 256, 0, stream>>>(xin, W, hlin, N, Kdim);
        // attention scalars
        attn_dots<<<blocks_for(nh * 32, 256), 256, 0, stream>>>(
            hlin, att_s, att_d, a_s, a_d, N);
        // init reductions
        fill_f32<<<blocks_for(nh, 256), 256, 0, stream>>>(emax, -__builtin_huge_valf(), nh);
        fill_f32<<<blocks_for(nh, 256), 256, 0, stream>>>(denom, 0.0f, nh);
        fill_f32<<<blocks_for(nhc, 256), 256, 0, stream>>>(agg, 0.0f, nhc);
        // segment softmax + scatter
        edge_max<<<blocks_for(etot, 256), 256, 0, stream>>>(ei, a_s, a_d, emax, E, N);
        edge_denom<<<blocks_for(etot, 256), 256, 0, stream>>>(ei, a_s, a_d, emax, denom, E, N);
        edge_scatter<<<blocks_for(etot * HEADS, 256), 256, 0, stream>>>(
            ei, a_s, a_d, emax, denom, hlin, agg, E, N);
        // bias + relu
        bias_relu<<<blocks_for(nhc, 256), 256, 0, stream>>>(agg, bias, hout, nhc);
    };

    layer(x,  Kin, W1, as1, ad1, b1, h1);
    layer(h1, HC,  W2, as2, ad2, b2, agg);   // layer-2 output lands in `agg`

    // pooling + head
    fill_f32<<<blocks_for((long)G * HC, 256), 256, 0, stream>>>(pooled, 0.0f, (long)G * HC);
    fill_f32<<<1, 256, 0, stream>>>(cnt, 0.0f, G);
    pool_accum<<<blocks_for((long)N * 4, 256), 256, 0, stream>>>(agg, batc, pooled, cnt, N);
    final_lin<<<blocks_for((long)G * 32, 256), 256, 0, stream>>>(pooled, cnt, lw, lb, out, G);
}
